// MSDALayer_20134806684287
// MI455X (gfx1250) — matrix operand order repaired and checked by reading, not compiled, not run on hardware
//
#include <hip/hip_runtime.h>
#include <math.h>

typedef __bf16 bf16_t;
typedef __attribute__((ext_vector_type(16))) __bf16 v16bf;
typedef __attribute__((ext_vector_type(8)))  float  v8f;

#define NTOK 32768
#define DIMC 256

// ---------------- workspace layout (bytes) ----------------
static constexpr size_t OFF_XN   = 0;            // xn packed-A bf16   16,777,216
static constexpr size_t OFF_QKV  = 16777216;     // qkv f32           100,663,296
static constexpr size_t OFF_ATTN = 117440512;    // attn packed-A bf16 16,777,216
static constexpr size_t OFF_MERG = 134217728;    // merged f32         33,554,432
static constexpr size_t OFF_ML   = 0;            // ml f32             33,554,432
static constexpr size_t OFF_MLBF = 33554432;     // ml packed-A bf16   16,777,216
static constexpr size_t OFF_HID  = 50331648;     // hid packed-A bf16  67,108,864
static constexpr size_t OFF_MO   = 167772160;    // mo f32             33,554,432
static constexpr size_t OFF_Y    = 0;            // y f32              33,554,432
static constexpr size_t OFF_X2   = 33554432;     // x2 f32             33,554,432
static constexpr size_t OFF_ST   = 67108864;     // stats (floats)
static constexpr size_t OFF_W    = 201326592;    // packed bf16 weights

// ---------------- packed fragment layouts (wave32 WMMA 16x16x32 bf16) ----------
// A fragment (mT,kb): lane = (row&15) + 16*(kk>>4); e = kk & 15
//   linear: ((mT*(K/32)+kb)*32 + lane)*16 + e        (kk = k & 31)
// B fragment (nT,kb): lane = (n&15) + 16*(kk>>4); e = kk & 15
//   linear: ((nT*(K/32)+kb)*32 + lane)*16 + e
__device__ __forceinline__ size_t packA_idx(int row, int k, int Kdim) {
  int mT = row >> 4, kb = k >> 5, kk = k & 31;
  int lane = (row & 15) + ((kk >> 4) << 4);
  int e = kk & 15;
  return (((size_t)mT * (Kdim >> 5) + kb) * 32 + lane) * 16 + e;
}

// 4 M-tiles per wave share one B fragment per K-step: 1 B load + 4 A loads + 4 WMMA
__device__ __forceinline__ void wmma4(const bf16_t* __restrict__ Apack,
                                      const bf16_t* __restrict__ Bpack,
                                      int mT0, int nT, int Kdim, int lane,
                                      v8f acc[4]) {
  const int kbn = Kdim >> 5;
  const bf16_t* bbase = Bpack + ((size_t)nT * kbn * 32 + lane) * 16;
  const bf16_t* abase = Apack + ((size_t)mT0 * kbn * 32 + lane) * 16;
  const size_t astride = (size_t)kbn * 512;       // next M-tile
  for (int kb = 0; kb < kbn; ++kb) {
    v16bf b = *(const v16bf*)(bbase + (size_t)kb * 512);
    __builtin_prefetch(bbase + (size_t)(kb + 1) * 512, 0, 1);  // global_prefetch_b8
#pragma unroll
    for (int m = 0; m < 4; ++m) {
      v16bf a = *(const v16bf*)(abase + (size_t)m * astride + (size_t)kb * 512);
      acc[m] = __builtin_amdgcn_wmma_f32_16x16x32_bf16(false, a, false, b,
                                                       (short)0, acc[m], false, false);
    }
  }
}

// ---------------- weight pre-swizzle into B-fragment layout ----------------
// srcTrans=0: W[k,n] = src[k*N+n] ; srcTrans=1: W[k,n] = src[n*K+k]
__global__ void k_pack_b(const float* __restrict__ src, bf16_t* __restrict__ dst,
                         int K, int N, int srcTrans) {
  int gid = blockIdx.x * 256 + threadIdx.x;
  if (gid >= K * N) return;
  int e = gid & 15;
  int l = (gid >> 4) & 31;
  int blk = gid >> 9;
  int kbn = K >> 5;
  int kb = blk % kbn, nT = blk / kbn;
  int n = nT * 16 + (l & 15);
  int k = kb * 32 + ((l >> 4) << 4) + e;
  float v = srcTrans ? src[(size_t)n * K + k] : src[(size_t)k * N + n];
  dst[gid] = (bf16_t)v;
}

// ---------------- LayerNorm 1 -> xn packed-A (K=256) ----------------
__global__ void k_ln1(const float* __restrict__ x, const float* __restrict__ g,
                      const float* __restrict__ bt, bf16_t* __restrict__ xnP) {
  int t = blockIdx.x, c = threadIdx.x;
  int b = t >> 12, hw = t & 4095;
  float v = x[(((size_t)b * DIMC + c) << 12) + hw];
  __shared__ float s1[256], s2[256];
  s1[c] = v; s2[c] = v * v; __syncthreads();
  for (int s = 128; s > 0; s >>= 1) {
    if (c < s) { s1[c] += s1[c + s]; s2[c] += s2[c + s]; }
    __syncthreads();
  }
  float mean = s1[0] * (1.0f / 256.0f);
  float var  = s2[0] * (1.0f / 256.0f) - mean * mean;
  float rstd = rsqrtf(var + 1e-5f);
  xnP[packA_idx(t, c, DIMC)] = (bf16_t)((v - mean) * rstd * g[c] + bt[c]);
}

// ---------------- QKV GEMM: 64x16 strip per wave -> qkv planes [o][t] ----------
__global__ void k_qkv_gemm(const bf16_t* __restrict__ xnP, const bf16_t* __restrict__ wP,
                           float* __restrict__ qkv) {
  int lane = threadIdx.x & 31;
  int wave = blockIdx.x * 8 + (threadIdx.x >> 5);
  int mG = wave & 511, nT = wave >> 9;            // 512 x 48
  v8f acc[4] = {};
  wmma4(xnP, wP, mG * 4, nT, DIMC, lane, acc);
  int col = nT * 16 + (lane & 15);
#pragma unroll
  for (int m = 0; m < 4; ++m) {
    int rbase = (mG * 4 + m) * 16 + ((lane >> 4) << 3);
#pragma unroll
    for (int i = 0; i < 8; ++i) qkv[(size_t)col * NTOK + rbase + i] = acc[m][i];
  }
}

// ---------------- dilated 3x3 attention -> attn packed-A (K=64, per branch) ----
__global__ void k_attn(const float* __restrict__ qkv, bf16_t* __restrict__ attnP) {
  int gid = blockIdx.x * 256 + threadIdx.x;
  int pix = gid & 4095;
  int rest = gid >> 12;
  int head = rest & 1; rest >>= 1;
  int b = rest & 7;
  int br = rest >> 3;
  int d = br + 1;
  int h = pix >> 6, w = pix & 63;
  size_t t = (size_t)b * 4096 + pix;
  const float* qp = qkv + ((size_t)(br * 64 + head * 32)) * NTOK + t;
  const float* kb = qkv + ((size_t)(256 + br * 64 + head * 32)) * NTOK + (size_t)b * 4096;
  const float* vb = qkv + ((size_t)(512 + br * 64 + head * 32)) * NTOK + (size_t)b * 4096;
  float q[32];
#pragma unroll
  for (int dd = 0; dd < 32; ++dd) q[dd] = qp[(size_t)dd * NTOK];
  const float scale = 0.17677669529663687f;  // 32^-0.5
  float logit[9];
#pragma unroll
  for (int n = 0; n < 9; ++n) {
    int ih = h + (n / 3 - 1) * d, iw = w + (n % 3 - 1) * d;
    float acc = 0.0f;
    if ((unsigned)ih < 64u && (unsigned)iw < 64u) {
      int np = ih * 64 + iw;
#pragma unroll
      for (int dd = 0; dd < 32; ++dd) acc += q[dd] * kb[(size_t)dd * NTOK + np];
      acc *= scale;
    }
    logit[n] = acc;  // zero-padded neighbors keep logit 0 (matches reference)
  }
  float m = logit[0];
#pragma unroll
  for (int n = 1; n < 9; ++n) m = fmaxf(m, logit[n]);
  float sum = 0.0f;
#pragma unroll
  for (int n = 0; n < 9; ++n) { logit[n] = expf(logit[n] - m); sum += logit[n]; }
  float inv = 1.0f / sum;
  float o[32];
#pragma unroll
  for (int dd = 0; dd < 32; ++dd) o[dd] = 0.0f;
#pragma unroll
  for (int n = 0; n < 9; ++n) {
    int ih = h + (n / 3 - 1) * d, iw = w + (n % 3 - 1) * d;
    if ((unsigned)ih < 64u && (unsigned)iw < 64u) {
      int np = ih * 64 + iw;
      float p = logit[n] * inv;
#pragma unroll
      for (int dd = 0; dd < 32; ++dd) o[dd] += p * vb[(size_t)dd * NTOK + np];
    }
  }
  bf16_t* op = attnP + (size_t)br * NTOK * 64;
#pragma unroll
  for (int dd = 0; dd < 32; ++dd)
    op[packA_idx((int)t, head * 32 + dd, 64)] = (bf16_t)o[dd];
}

// ---------------- branch projection GEMM + bias + residual -> merged -----------
__global__ void k_proj_gemm(const bf16_t* __restrict__ attnP, const bf16_t* __restrict__ projP,
                            const float* __restrict__ proj_b, const float* __restrict__ x,
                            float* __restrict__ merged) {
  int br = blockIdx.y;
  int lane = threadIdx.x & 31;
  int wave = blockIdx.x * 8 + (threadIdx.x >> 5);
  int mG = wave & 511, nT = wave >> 9;            // 512 x 4
  const bf16_t* A = attnP + (size_t)br * NTOK * 64;
  const bf16_t* Bm = projP + (size_t)br * 64 * 64;
  v8f acc[4] = {};
  wmma4(A, Bm, mG * 4, nT, 64, lane, acc);
  int col = nT * 16 + (lane & 15);
  float bias = proj_b[br * 64 + col];
  int ch = br * 64 + col;
#pragma unroll
  for (int m = 0; m < 4; ++m) {
    int rbase = (mG * 4 + m) * 16 + ((lane >> 4) << 3);
#pragma unroll
    for (int i = 0; i < 8; ++i) {
      int t = rbase + i;
      int b = t >> 12, hw = t & 4095;
      size_t idx = (((size_t)b * DIMC + ch) << 12) + hw;
      merged[idx] = acc[m][i] + bias + x[idx];
    }
  }
}

// ---------------- LayerNorm 2 -> ml f32 (token-major) + packed-A bf16 ----------
__global__ void k_ln2(const float* __restrict__ merged, const float* __restrict__ g,
                      const float* __restrict__ bt, float* __restrict__ ml,
                      bf16_t* __restrict__ mlP) {
  int t = blockIdx.x, c = threadIdx.x;
  int b = t >> 12, hw = t & 4095;
  float v = merged[(((size_t)b * DIMC + c) << 12) + hw];
  __shared__ float s1[256], s2[256];
  s1[c] = v; s2[c] = v * v; __syncthreads();
  for (int s = 128; s > 0; s >>= 1) {
    if (c < s) { s1[c] += s1[c + s]; s2[c] += s2[c + s]; }
    __syncthreads();
  }
  float mean = s1[0] * (1.0f / 256.0f);
  float var  = s2[0] * (1.0f / 256.0f) - mean * mean;
  float rstd = rsqrtf(var + 1e-5f);
  float o = (v - mean) * rstd * g[c] + bt[c];
  ml[(size_t)t * DIMC + c] = o;
  mlP[packA_idx(t, c, DIMC)] = (bf16_t)o;
}

// ---------------- MLP GEMM1 (256->1024) + exact GELU -> hid packed-A (K=1024) --
__global__ void k_mlp1(const bf16_t* __restrict__ mlP, const bf16_t* __restrict__ w1P,
                       const float* __restrict__ b1, bf16_t* __restrict__ hidP) {
  int lane = threadIdx.x & 31;
  int wave = blockIdx.x * 8 + (threadIdx.x >> 5);
  int mG = wave & 511, nT = wave >> 9;            // 512 x 64
  v8f acc[4] = {};
  wmma4(mlP, w1P, mG * 4, nT, DIMC, lane, acc);
  int col = nT * 16 + (lane & 15);
  float bias = b1[col];
#pragma unroll
  for (int m = 0; m < 4; ++m) {
    int rbase = (mG * 4 + m) * 16 + ((lane >> 4) << 3);
#pragma unroll
    for (int i = 0; i < 8; ++i) {
      float v = acc[m][i] + bias;
      float gl = 0.5f * v * (1.0f + erff(v * 0.7071067811865475f));
      hidP[packA_idx(rbase + i, col, 1024)] = (bf16_t)gl;
    }
  }
}

// ---------------- MLP GEMM2 (1024->256) + bias + residual(ml) -> mo planes -----
__global__ void k_mlp2(const bf16_t* __restrict__ hidP, const bf16_t* __restrict__ w2P,
                       const float* __restrict__ b2, const float* __restrict__ ml,
                       float* __restrict__ mo) {
  int lane = threadIdx.x & 31;
  int wave = blockIdx.x * 8 + (threadIdx.x >> 5);
  int mG = wave & 511, nT = wave >> 9;            // 512 x 16
  v8f acc[4] = {};
  wmma4(hidP, w2P, mG * 4, nT, 1024, lane, acc);
  int col = nT * 16 + (lane & 15);
  float bias = b2[col];
#pragma unroll
  for (int m = 0; m < 4; ++m) {
    int rbase = (mG * 4 + m) * 16 + ((lane >> 4) << 3);
#pragma unroll
    for (int i = 0; i < 8; ++i) {
      int t = rbase + i;
      int b = t >> 12, hw = t & 4095;
      mo[(((size_t)b * DIMC + col) << 12) + hw] =
          acc[m][i] + bias + ml[(size_t)t * DIMC + col];
    }
  }
}

// ---------------- EMA: row/col pooling ----------------
__global__ void k_pool(const float* __restrict__ mo, float* __restrict__ x_h,
                       float* __restrict__ x_w) {
  int blk = blockIdx.x;           // bg*32 + c
  int c = blk & 31, bg = blk >> 5;
  int b = bg >> 3, gr = bg & 7;
  const float* pl = mo + (((size_t)b * DIMC + gr * 32 + c) << 12);
  int j = threadIdx.x;            // 64
  float sh = 0.0f, sw = 0.0f;
  for (int k = 0; k < 64; ++k) { sh += pl[j * 64 + k]; sw += pl[k * 64 + j]; }
  x_h[(size_t)blk * 64 + j] = sh * (1.0f / 64.0f);
  x_w[(size_t)blk * 64 + j] = sw * (1.0f / 64.0f);
}

// ---------------- EMA: 1x1 over pooled ----------------
__global__ void k_c1(const float* __restrict__ x_h, const float* __restrict__ x_w,
                     const float* __restrict__ w, const float* __restrict__ bias,
                     float* __restrict__ xh2, float* __restrict__ xw2) {
  int bg = blockIdx.x;
  int xy = threadIdx.x;           // 128: 0..63 = h rows, 64..127 = w cols
  __shared__ float sw[1024];
  for (int i = threadIdx.x; i < 1024; i += 128) sw[i] = w[i];
  __syncthreads();
  float in[32];
  int pos = xy & 63;
  bool isH = xy < 64;
#pragma unroll
  for (int c = 0; c < 32; ++c)
    in[c] = isH ? x_h[(size_t)(bg * 32 + c) * 64 + pos]
                : x_w[(size_t)(bg * 32 + c) * 64 + pos];
  for (int o = 0; o < 32; ++o) {
    float acc = bias[o];
#pragma unroll
    for (int c = 0; c < 32; ++c) acc += sw[o * 32 + c] * in[c];
    if (isH) xh2[(size_t)(bg * 32 + o) * 64 + pos] = acc;
    else     xw2[(size_t)(bg * 32 + o) * 64 + pos] = acc;
  }
}

// ---------------- EMA: y = gx*sig(xh)*sig(xw), per-(bg,c) stats ----------------
__global__ void k_ygate(const float* __restrict__ mo, const float* __restrict__ xh2,
                        const float* __restrict__ xw2, float* __restrict__ y,
                        float* __restrict__ mu, float* __restrict__ rstd,
                        float* __restrict__ x1m, const float* __restrict__ gn_b) {
  int blk = blockIdx.x;           // bg*32 + c
  int c = blk & 31, bg = blk >> 5;
  int b = bg >> 3, gr = bg & 7;
  const float* pl = mo + (((size_t)b * DIMC + gr * 32 + c) << 12);
  __shared__ float s1[256], s2[256];
  float ls = 0.0f, lq = 0.0f;
  for (int it = 0; it < 16; ++it) {
    int pix = it * 256 + threadIdx.x;
    int h = pix >> 6, wv = pix & 63;
    float gh = 1.0f / (1.0f + expf(-xh2[(size_t)blk * 64 + h]));
    float gw = 1.0f / (1.0f + expf(-xw2[(size_t)blk * 64 + wv]));
    float val = pl[pix] * gh * gw;
    y[((size_t)blk << 12) + pix] = val;
    ls += val; lq += val * val;
  }
  s1[threadIdx.x] = ls; s2[threadIdx.x] = lq; __syncthreads();
  for (int s = 128; s > 0; s >>= 1) {
    if (threadIdx.x < s) { s1[threadIdx.x] += s1[threadIdx.x + s];
                           s2[threadIdx.x] += s2[threadIdx.x + s]; }
    __syncthreads();
  }
  if (threadIdx.x == 0) {
    float m = s1[0] * (1.0f / 4096.0f);
    float v = s2[0] * (1.0f / 4096.0f) - m * m;
    mu[blk] = m;
    rstd[blk] = rsqrtf(v + 1e-5f);
    x1m[blk] = gn_b[c];  // mean of groupnorm output == beta (exact)
  }
}

// ---------------- EMA: 3x3 conv on gx -> x2 ----------------
__global__ void k_conv3(const float* __restrict__ mo, const float* __restrict__ w,
                        const float* __restrict__ bias, float* __restrict__ x2) {
  int blk = blockIdx.x;           // bg(6b) | o(5b) | chunk(4b)
  int chunk = blk & 15;
  int o = (blk >> 4) & 31;
  int bg = blk >> 9;
  int b = bg >> 3, gr = bg & 7;
  __shared__ float sw[288];
  for (int i = threadIdx.x; i < 288; i += 256) sw[i] = w[o * 288 + i];
  __syncthreads();
  int pix = chunk * 256 + threadIdx.x;
  int h = pix >> 6, wc = pix & 63;
  const float* gx = mo + (((size_t)b * DIMC + gr * 32) << 12);
  float acc = bias[o];
  for (int c = 0; c < 32; ++c) {
    const float* pl = gx + ((size_t)c << 12);
#pragma unroll
    for (int ki = 0; ki < 3; ++ki) {
      int ih = h + ki - 1;
      if ((unsigned)ih >= 64u) continue;
#pragma unroll
      for (int kj = 0; kj < 3; ++kj) {
        int iw = wc + kj - 1;
        if ((unsigned)iw >= 64u) continue;
        acc += sw[c * 9 + ki * 3 + kj] * pl[ih * 64 + iw];
      }
    }
  }
  x2[(((size_t)bg * 32 + o) << 12) + pix] = acc;
}

// ---------------- EMA: mean of x2 per (bg,c) ----------------
__global__ void k_x2mean(const float* __restrict__ x2, float* __restrict__ x2m) {
  int blk = blockIdx.x;
  __shared__ float s1[256];
  float ls = 0.0f;
  for (int it = 0; it < 16; ++it) ls += x2[((size_t)blk << 12) + it * 256 + threadIdx.x];
  s1[threadIdx.x] = ls; __syncthreads();
  for (int s = 128; s > 0; s >>= 1) {
    if (threadIdx.x < s) s1[threadIdx.x] += s1[threadIdx.x + s];
    __syncthreads();
  }
  if (threadIdx.x == 0) x2m[blk] = s1[0] * (1.0f / 4096.0f);
}

// ---------------- EMA: channel softmaxes ----------------
__global__ void k_smax(const float* __restrict__ x1m, const float* __restrict__ x2m,
                       float* __restrict__ x11, float* __restrict__ x21) {
  int bg = blockIdx.x, c = threadIdx.x;  // 32 threads
  __shared__ float a[32], d[32];
  a[c] = x1m[bg * 32 + c]; d[c] = x2m[bg * 32 + c]; __syncthreads();
  float m1 = -1e30f, m2 = -1e30f;
  for (int i = 0; i < 32; ++i) { m1 = fmaxf(m1, a[i]); m2 = fmaxf(m2, d[i]); }
  float s1 = 0.0f, s2 = 0.0f;
  for (int i = 0; i < 32; ++i) { s1 += expf(a[i] - m1); s2 += expf(d[i] - m2); }
  x11[bg * 32 + c] = expf(a[c] - m1) / s1;
  x21[bg * 32 + c] = expf(d[c] - m2) / s2;
}

// ---------------- EMA final gate + channel-split stack -> out(4,B,64,H,W) ------
__global__ void k_final(const float* __restrict__ mo, const float* __restrict__ y,
                        const float* __restrict__ x2, const float* __restrict__ mu,
                        const float* __restrict__ rstd, const float* __restrict__ x11,
                        const float* __restrict__ x21, const float* __restrict__ gn_g,
                        const float* __restrict__ gn_b, float* __restrict__ out) {
  int bg = blockIdx.x >> 4;                       // 16 blocks per bg
  int pix = ((blockIdx.x & 15) << 8) + threadIdx.x;
  int b = bg >> 3, gr = bg & 7;
  __shared__ float s11[32], s21[32], smu[32], srs[32], sg[32], sb[32];
  if (threadIdx.x < 32) {
    int c = threadIdx.x;
    s11[c] = x11[bg * 32 + c]; s21[c] = x21[bg * 32 + c];
    smu[c] = mu[bg * 32 + c];  srs[c] = rstd[bg * 32 + c];
    sg[c] = gn_g[c]; sb[c] = gn_b[c];
  }
  __syncthreads();
  float wsum = 0.0f;
#pragma unroll 4
  for (int c = 0; c < 32; ++c) {
    size_t idx = (((size_t)bg * 32 + c) << 12) + pix;
    float x1v = (y[idx] - smu[c]) * srs[c] * sg[c] + sb[c];
    wsum += s11[c] * x2[idx] + s21[c] * x1v;
  }
  float sig = 1.0f / (1.0f + expf(-wsum));
#pragma unroll 4
  for (int c = 0; c < 32; ++c) {
    int C = gr * 32 + c;
    int br = C >> 6, cb = C & 63;
    float gxv = mo[(((size_t)b * DIMC + C) << 12) + pix];
    out[((((size_t)br * 8 + b) * 64 + cb) << 12) + pix] = gxv * sig;
  }
}

// ---------------- launch ----------------
extern "C" void kernel_launch(void* const* d_in, const int* in_sizes, int n_in,
                              void* d_out, int out_size, void* d_ws, size_t ws_size,
                              hipStream_t stream) {
  (void)in_sizes; (void)n_in; (void)out_size; (void)ws_size;
  const float* x       = (const float*)d_in[0];
  const float* n1g     = (const float*)d_in[1];
  const float* n1b     = (const float*)d_in[2];
  const float* qkv_w   = (const float*)d_in[3];
  const float* proj_w  = (const float*)d_in[4];
  const float* proj_b  = (const float*)d_in[5];
  const float* n2g     = (const float*)d_in[6];
  const float* n2b     = (const float*)d_in[7];
  const float* mlp_w1  = (const float*)d_in[8];
  const float* mlp_b1  = (const float*)d_in[9];
  const float* mlp_w2  = (const float*)d_in[10];
  const float* mlp_b2  = (const float*)d_in[11];
  const float* c1_w    = (const float*)d_in[12];
  const float* c1_b    = (const float*)d_in[13];
  const float* c3_w    = (const float*)d_in[14];
  const float* c3_b    = (const float*)d_in[15];
  const float* gn_g    = (const float*)d_in[16];
  const float* gn_b    = (const float*)d_in[17];
  float* out = (float*)d_out;

  char* ws = (char*)d_ws;
  bf16_t* xnP    = (bf16_t*)(ws + OFF_XN);
  float*  qkv    = (float*) (ws + OFF_QKV);
  bf16_t* attnP  = (bf16_t*)(ws + OFF_ATTN);
  float*  merged = (float*) (ws + OFF_MERG);
  float*  ml     = (float*) (ws + OFF_ML);
  bf16_t* mlP    = (bf16_t*)(ws + OFF_MLBF);
  bf16_t* hidP   = (bf16_t*)(ws + OFF_HID);
  float*  mo     = (float*) (ws + OFF_MO);
  float*  yb     = (float*) (ws + OFF_Y);
  float*  x2b    = (float*) (ws + OFF_X2);
  float*  st     = (float*) (ws + OFF_ST);
  float* x_h  = st;             float* x_w  = st + 131072;
  float* xh2  = st + 262144;    float* xw2  = st + 393216;
  float* muA  = st + 524288;    float* rsA  = st + 526336;
  float* x1m  = st + 528384;    float* x2m  = st + 530432;
  float* x11  = st + 532480;    float* x21  = st + 534528;
  bf16_t* qkvP  = (bf16_t*)(ws + OFF_W);   // 196608
  bf16_t* w1P   = qkvP + 196608;           // 262144
  bf16_t* w2P   = w1P + 262144;            // 262144
  bf16_t* projP = w2P + 262144;            // 16384

  // weights -> bf16 B-fragment layout
  k_pack_b<<<768, 256, 0, stream>>>(qkv_w, qkvP, 256, 768, 1);
  k_pack_b<<<1024, 256, 0, stream>>>(mlp_w1, w1P, 256, 1024, 0);
  k_pack_b<<<1024, 256, 0, stream>>>(mlp_w2, w2P, 1024, 256, 0);
  for (int i = 0; i < 4; ++i)
    k_pack_b<<<16, 256, 0, stream>>>(proj_w + i * 4096, projP + i * 4096, 64, 64, 1);

  k_ln1<<<NTOK, 256, 0, stream>>>(x, n1g, n1b, xnP);
  k_qkv_gemm<<<3072, 256, 0, stream>>>(xnP, qkvP, qkv);                // 512 x 48
  k_attn<<<1024, 256, 0, stream>>>(qkv, attnP);
  k_proj_gemm<<<dim3(256, 4), 256, 0, stream>>>(attnP, projP, proj_b, x, merged);
  k_ln2<<<NTOK, 256, 0, stream>>>(merged, n2g, n2b, ml, mlP);
  k_mlp1<<<4096, 256, 0, stream>>>(mlP, w1P, mlp_b1, hidP);            // 512 x 64
  k_mlp2<<<1024, 256, 0, stream>>>(hidP, w2P, mlp_b2, ml, mo);         // 512 x 16

  k_pool<<<2048, 64, 0, stream>>>(mo, x_h, x_w);
  k_c1<<<64, 128, 0, stream>>>(x_h, x_w, c1_w, c1_b, xh2, xw2);
  k_ygate<<<2048, 256, 0, stream>>>(mo, xh2, xw2, yb, muA, rsA, x1m, gn_b);
  k_conv3<<<32768, 256, 0, stream>>>(mo, c3_w, c3_b, x2b);
  k_x2mean<<<2048, 256, 0, stream>>>(x2b, x2m);
  k_smax<<<64, 32, 0, stream>>>(x1m, x2m, x11, x21);
  k_final<<<1024, 256, 0, stream>>>(mo, yb, x2b, muA, rsA, x11, x21, gn_g, gn_b, out);
}
